// MultiHeadAttention_12919261626822
// MI455X (gfx1250) — matrix operand order repaired and checked by reading, not compiled, not run on hardware
//
#include <hip/hip_runtime.h>

#define B_   2
#define SQ_  2048
#define SK_  2048
#define E_   1024
#define H_   16
#define D_   64
#define NEG_INF (-3.4028234663852886e+38f)

typedef _Float16 half8  __attribute__((ext_vector_type(8)));
typedef _Float16 half16 __attribute__((ext_vector_type(16)));
typedef float    f32x8  __attribute__((ext_vector_type(8)));
typedef __attribute__((__vector_size__(4 * sizeof(int)))) int i32x4;

// ---- CDNA5 async global->LDS copy (ASYNCcnt path), with safe fallback -----
#if defined(__has_builtin)
#if __has_builtin(__builtin_amdgcn_global_load_async_to_lds_b128) && \
    __has_builtin(__builtin_amdgcn_s_wait_asynccnt)
#define HAVE_ASYNC_LDS 1
#endif
#if __has_builtin(__builtin_amdgcn_ds_swizzle)
#define HAVE_DS_SWIZZLE 1
#endif
#endif
#ifndef HAVE_ASYNC_LDS
#define HAVE_ASYNC_LDS 0
#endif
#ifndef HAVE_DS_SWIZZLE
#define HAVE_DS_SWIZZLE 0
#endif

typedef __attribute__((address_space(1))) i32x4 gas_i32x4;  // global
typedef __attribute__((address_space(3))) i32x4 las_i32x4;  // LDS

// Copy 16B from global to LDS (per lane).
static __device__ __forceinline__ void cp16B(const _Float16* g, _Float16* l) {
#if HAVE_ASYNC_LDS
  __builtin_amdgcn_global_load_async_to_lds_b128(
      (gas_i32x4*)g, (las_i32x4*)l, 0, 0);
#else
  *(half8*)l = *(const half8*)g;
#endif
}
static __device__ __forceinline__ void cp_fence_lds() {
#if HAVE_ASYNC_LDS
  __builtin_amdgcn_s_wait_asynccnt(0);  // async writes visible before barrier
#endif
}

// XOR-butterfly lane exchange within wave32 via ds_swizzle (no bpermute cruft).
template <int XM>
static __device__ __forceinline__ float swz_xor(float v) {
#if HAVE_DS_SWIZZLE
  return __int_as_float(
      __builtin_amdgcn_ds_swizzle(__float_as_int(v), (XM << 10) | 0x1f));
#else
  return __shfl_xor(v, XM, 32);
#endif
}

static __device__ __forceinline__ half16 frag_cat(half8 lo, half8 hi) {
  return __builtin_shufflevector(lo, hi, 0,1,2,3,4,5,6,7,8,9,10,11,12,13,14,15);
}

static __device__ __forceinline__ f32x8 wmma_f16(half16 a, half16 b, f32x8 c) {
  // D = A(16x32 f16) * B(32x16 f16) + C(16x16 f32)
  return __builtin_amdgcn_wmma_f32_16x16x32_f16(false, a, false, b, (short)0, c,
                                                false, false);
}

// Stage 16 contiguous elements of one row into LDS as f16 (two 16B chunks).
static __device__ __forceinline__ void stage16(const float* __restrict__ src,
                                               _Float16* dst) {
  const float4* s = (const float4*)src;
  float4 f0 = s[0], f1 = s[1], f2 = s[2], f3 = s[3];
  half8 h0 = { (_Float16)f0.x, (_Float16)f0.y, (_Float16)f0.z, (_Float16)f0.w,
               (_Float16)f1.x, (_Float16)f1.y, (_Float16)f1.z, (_Float16)f1.w };
  half8 h1 = { (_Float16)f2.x, (_Float16)f2.y, (_Float16)f2.z, (_Float16)f2.w,
               (_Float16)f3.x, (_Float16)f3.y, (_Float16)f3.z, (_Float16)f3.w };
  *(half8*)dst = h0;
  *(half8*)(dst + 8) = h1;
}
static __device__ __forceinline__ void stage16(const _Float16* __restrict__ src,
                                               _Float16* dst) {
  cp16B(src, dst);          // async path when available
  cp16B(src + 8, dst + 8);
}

// ---------------------------------------------------------------------------
// C[M,N] = A[M,K] @ W[N,K]^T + bias   (torch Linear, NT layout: both operands
// K-contiguous -> ideal for WMMA fragments).  Tile 128x128, 8 waves of 32x64.
// ---------------------------------------------------------------------------
template <typename AT, bool HALF_OUT>
__global__ void __launch_bounds__(256) gemm_nt_kernel(
    const AT* __restrict__ A, const float* __restrict__ W,
    const float* __restrict__ bias, void* __restrict__ Cout,
    int M, int N, int K)
{
  __shared__ _Float16 lA[128 * 40];  // 128 rows x 32 K halves, stride 40
  __shared__ _Float16 lB[128 * 40];

  const int t    = threadIdx.x;
  const int wave = t >> 5, lane = t & 31, ln = lane & 15, g = lane >> 4;
  const int wr   = wave >> 1;          // 0..3 -> 32-row strip
  const int wc   = wave & 1;           // 0..1 -> 64-col strip
  const int bm   = blockIdx.x, bn = blockIdx.y;

  f32x8 acc[2][4] = {};

  const int row = t >> 1;              // 0..127 : staging row
  const int c0  = (t & 1) * 16;        // staging column offset
  const int nk  = K >> 5;

  for (int kb = 0; kb < nk; ++kb) {
    __syncthreads();
    stage16(A + (size_t)(bm * 128 + row) * K + kb * 32 + c0, lA + row * 40 + c0);
    stage16(W + (size_t)(bn * 128 + row) * K + kb * 32 + c0, lB + row * 40 + c0);
    if (kb + 1 < nk) {  // gfx1250 global_prefetch_b8 for next K tile
      __builtin_prefetch(A + (size_t)(bm * 128 + row) * K + kb * 32 + 32 + c0, 0, 1);
      __builtin_prefetch(W + (size_t)(bn * 128 + row) * K + kb * 32 + 32 + c0, 0, 1);
    }
    cp_fence_lds();
    __syncthreads();

    half16 af[2], bf[4];
#pragma unroll
    for (int mt = 0; mt < 2; ++mt) {
      const _Float16* p = lA + (wr * 32 + mt * 16 + ln) * 40;
      // A 16x32 layout: halves 0..7 -> K = 8g..8g+7, halves 8..15 -> K = 16+8g..
      af[mt] = frag_cat(*(const half8*)(p + 8 * g), *(const half8*)(p + 16 + 8 * g));
    }
#pragma unroll
    for (int nt = 0; nt < 4; ++nt) {
      const _Float16* p = lB + (wc * 64 + nt * 16 + ln) * 40;
      // B 32x16 layout: lane holds column n, halves 0..7 -> K = 8g..8g+7, halves 8..15 -> K = 16+8g..
      bf[nt] = frag_cat(*(const half8*)(p + 8 * g), *(const half8*)(p + 16 + 8 * g));
    }
#pragma unroll
    for (int mt = 0; mt < 2; ++mt)
#pragma unroll
      for (int nt = 0; nt < 4; ++nt)
        acc[mt][nt] = wmma_f16(af[mt], bf[nt], acc[mt][nt]);
  }

#pragma unroll
  for (int mt = 0; mt < 2; ++mt) {
#pragma unroll
    for (int nt = 0; nt < 4; ++nt) {
      const int col = bn * 128 + wc * 64 + nt * 16 + ln;
      const float bv = bias[col];
#pragma unroll
      for (int r = 0; r < 8; ++r) {
        const int grow = bm * 128 + wr * 32 + mt * 16 + r + 8 * g;  // C row = r + 8g
        const float v = acc[mt][nt][r] + bv;
        if (HALF_OUT)
          ((_Float16*)Cout)[(size_t)grow * N + col] = (_Float16)v;
        else
          ((float*)Cout)[(size_t)grow * N + col] = v;
      }
    }
  }
}

// ---------------------------------------------------------------------------
// Scores + softmax.  Block = (qb, h, b): 128 q rows.  Wave owns 16 q rows.
// Phase 1: S = Q K^T / 8 (WMMA), masks -> raw masked scores parked in attnW.
//          Per-LANE online (m,l) -- no cross-lane traffic in the hot loop.
// Merge:   one 4-step xor butterfly (ds_swizzle) per row at the end.
// Phase 2: normalize in place; zero-fill causally dead tiles.
// ---------------------------------------------------------------------------
__global__ void __launch_bounds__(256) attn_scores_softmax_kernel(
    const _Float16* __restrict__ Qh, const _Float16* __restrict__ Kh,
    const unsigned char* __restrict__ kpm, const unsigned char* __restrict__ amask,
    float* __restrict__ attnW)
{
  __shared__ _Float16 lK[128 * 72];  // 128 k-tokens x 64 d halves, stride 72

  const int t    = threadIdx.x;
  const int wave = t >> 5, lane = t & 31, ln = lane & 15, g = lane >> 4;
  const int qb   = blockIdx.x, h = blockIdx.y, b = blockIdx.z;
  const int qw   = qb * 128 + wave * 16;  // wave's first q row

  // Q fragments, loaded once, reused for every k tile (A layout direct from global)
  const size_t qoff = ((size_t)(b * SQ_ + qw + ln)) * E_ + h * D_;
  const half16 aq0 = frag_cat(*(const half8*)(Qh + qoff + 8 * g),
                              *(const half8*)(Qh + qoff + 16 + 8 * g));
  const half16 aq1 = frag_cat(*(const half8*)(Qh + qoff + 32 + 8 * g),
                              *(const half8*)(Qh + qoff + 48 + 8 * g));

  float mrow[8], lrow[8];
#pragma unroll
  for (int r = 0; r < 8; ++r) { mrow[r] = NEG_INF; lrow[r] = 0.f; }

  const size_t abase = ((size_t)(b * H_ + h)) * SQ_ * SK_;
  const int tok = t >> 1;
  const int c0  = (t & 1) * 32;

  for (int kb = 0; kb <= qb; ++kb) {            // causal: no k block beyond q block
    __syncthreads();
    const size_t koff = ((size_t)(b * SK_ + kb * 128 + tok)) * E_ + h * D_ + c0;
    cp16B(Kh + koff +  0, lK + tok * 72 + c0 +  0);
    cp16B(Kh + koff +  8, lK + tok * 72 + c0 +  8);
    cp16B(Kh + koff + 16, lK + tok * 72 + c0 + 16);
    cp16B(Kh + koff + 24, lK + tok * 72 + c0 + 24);
    cp_fence_lds();
    __syncthreads();

    for (int kt = 0; kt < 8; ++kt) {
      const int kstart = kb * 128 + kt * 16;
      if (kstart > qw + 15) break;              // wave-uniform: EXEC stays full for WMMA

      const _Float16* bp = lK + (kt * 16 + ln) * 72;
      const half16 b0 = frag_cat(*(const half8*)(bp + 8 * g),
                                 *(const half8*)(bp + 16 + 8 * g));
      const half16 b1 = frag_cat(*(const half8*)(bp + 32 + 8 * g),
                                 *(const half8*)(bp + 48 + 8 * g));
      f32x8 acc = {};
      acc = wmma_f16(aq0, b0, acc);
      acc = wmma_f16(aq1, b1, acc);

      const int kg = kstart + ln;
      const bool kp = kpm[b * SK_ + kg] != 0;
#pragma unroll
      for (int r = 0; r < 8; ++r) {
        const int q = qw + r + 8 * g;
        const bool msk = kp | (amask[(size_t)q * SK_ + kg] != 0);
        const float s = msk ? NEG_INF : acc[r] * 0.125f;  // 1/sqrt(64)
        attnW[abase + (size_t)q * SK_ + kg] = s;          // park raw score
        // per-lane online softmax state (no cross-lane ops here)
        const float mnew = fmaxf(mrow[r], s);
        const float e = msk ? 0.f : __expf(s - mnew);
        lrow[r] = lrow[r] * __expf(mrow[r] - mnew) + e;
        mrow[r] = mnew;
      }
    }
  }

  // Merge (m,l) across the 16 lanes of each C-fragment column group.
  float rl[8];
#pragma unroll
  for (int r = 0; r < 8; ++r) {
    float m = mrow[r], l = lrow[r];
#define MERGE_STEP(XM)                                     \
    {                                                      \
      const float om = swz_xor<XM>(m);                     \
      const float ol = swz_xor<XM>(l);                     \
      const float mn = fmaxf(m, om);                       \
      l = l * __expf(m - mn) + ol * __expf(om - mn);       \
      m = mn;                                              \
    }
    MERGE_STEP(1) MERGE_STEP(2) MERGE_STEP(4) MERGE_STEP(8)
#undef MERGE_STEP
    mrow[r] = m;
    rl[r] = 1.0f / l;
  }

  // Phase 2: normalize in place (same wave wrote these addresses -> ordered).
  for (int ktile = 0; ktile < SK_ / 16; ++ktile) {
    const int kg = ktile * 16 + ln;
    const bool live = (ktile * 16 <= qw + 15);
#pragma unroll
    for (int r = 0; r < 8; ++r) {
      const int q = qw + r + 8 * g;
      const size_t idx = abase + (size_t)q * SK_ + kg;
      float p = 0.f;
      if (live) {
        const float s = attnW[idx];
        p = (s == NEG_INF) ? 0.f : __expf(s - mrow[r]) * rl[r];
      }
      attnW[idx] = p;
    }
  }
}

// ---------------------------------------------------------------------------
// Context = P @ V per (b,h,128-q block).  P fp32 -> f16 in LDS (A frags),
// V staged transposed into LDS (B frags).  Wave owns 16 q rows x 64 d.
// ---------------------------------------------------------------------------
__global__ void __launch_bounds__(256) attn_pv_kernel(
    const float* __restrict__ attnW, const _Float16* __restrict__ Vh,
    _Float16* __restrict__ Ctx)
{
  __shared__ _Float16 lP[128 * 40];  // 128 q x 32 k
  __shared__ _Float16 lV[64 * 40];   // 64 d x 32 k (transposed)

  const int t    = threadIdx.x;
  const int wave = t >> 5, lane = t & 31, ln = lane & 15, g = lane >> 4;
  const int qb   = blockIdx.x, h = blockIdx.y, b = blockIdx.z;

  const size_t abase = ((size_t)(b * H_ + h)) * SQ_ * SK_;
  f32x8 acc[4] = {};

  const int nkb  = qb * 4 + 4;        // causal: k <= qb*128+127 -> (qb*4+4) 32-wide steps
  const int prow = t >> 1, pc0 = (t & 1) * 16;
  const int vtok = t >> 3, vd0 = (t & 7) * 8;

  for (int kb = 0; kb < nkb; ++kb) {
    __syncthreads();
    // stage P (fp32 -> f16)
    {
      const float4* s = (const float4*)(attnW + abase
                          + (size_t)(qb * 128 + prow) * SK_ + kb * 32 + pc0);
      float4 f0 = s[0], f1 = s[1], f2 = s[2], f3 = s[3];
      half8 h0 = { (_Float16)f0.x, (_Float16)f0.y, (_Float16)f0.z, (_Float16)f0.w,
                   (_Float16)f1.x, (_Float16)f1.y, (_Float16)f1.z, (_Float16)f1.w };
      half8 h1 = { (_Float16)f2.x, (_Float16)f2.y, (_Float16)f2.z, (_Float16)f2.w,
                   (_Float16)f3.x, (_Float16)f3.y, (_Float16)f3.z, (_Float16)f3.w };
      *(half8*)(lP + prow * 40 + pc0) = h0;
      *(half8*)(lP + prow * 40 + pc0 + 8) = h1;
    }
    // stage V transposed
    {
      const half8 vv = *(const half8*)(Vh + ((size_t)(b * SK_ + kb * 32 + vtok)) * E_
                                          + h * D_ + vd0);
#pragma unroll
      for (int i = 0; i < 8; ++i) lV[(vd0 + i) * 40 + vtok] = vv[i];  // transpose
    }
    __syncthreads();

    const _Float16* ap = lP + (wave * 16 + ln) * 40;
    const half16 a = frag_cat(*(const half8*)(ap + 8 * g),
                              *(const half8*)(ap + 16 + 8 * g));
#pragma unroll
    for (int nt = 0; nt < 4; ++nt) {
      const _Float16* bp = lV + (nt * 16 + ln) * 40;
      const half16 bfrag = frag_cat(*(const half8*)(bp + 8 * g),
                                    *(const half8*)(bp + 16 + 8 * g));
      acc[nt] = wmma_f16(a, bfrag, acc[nt]);
    }
  }

#pragma unroll
  for (int nt = 0; nt < 4; ++nt)
#pragma unroll
    for (int r = 0; r < 8; ++r) {
      const int q = qb * 128 + wave * 16 + r + 8 * g;
      Ctx[((size_t)(b * SQ_ + q)) * E_ + h * D_ + nt * 16 + ln] =
          (_Float16)acc[nt][r];
    }
}

// ---------------------------------------------------------------------------
extern "C" void kernel_launch(void* const* d_in, const int* in_sizes, int n_in,
                              void* d_out, int out_size, void* d_ws, size_t ws_size,
                              hipStream_t stream)
{
  (void)in_sizes; (void)n_in; (void)out_size; (void)ws_size;

  const float* query = (const float*)d_in[0];
  const float* key_  = (const float*)d_in[1];
  const float* value = (const float*)d_in[2];
  const unsigned char* kpm = (const unsigned char*)d_in[3];  // [B,SK] bool
  const unsigned char* am  = (const unsigned char*)d_in[4];  // [SQ,SK] bool
  const float* Wq = (const float*)d_in[5];
  const float* bq = (const float*)d_in[6];
  const float* Wk = (const float*)d_in[7];
  const float* bk = (const float*)d_in[8];
  const float* Wv = (const float*)d_in[9];
  const float* bv = (const float*)d_in[10];
  const float* Wo = (const float*)d_in[11];
  const float* bo = (const float*)d_in[12];

  float* out   = (float*)d_out;
  float* attnW = out + (size_t)B_ * SQ_ * E_;   // outputs concatenated: out, attn_weights

  const size_t tokE = (size_t)B_ * SQ_ * E_;    // 4M elements
  _Float16* Qh  = (_Float16*)d_ws;              // ws: Q,K,V,Ctx in f16 (32 MB)
  _Float16* Kh  = Qh + tokE;
  _Float16* Vh  = Kh + tokE;
  _Float16* Ctx = Vh + tokE;

  dim3 blk(256);
  dim3 gproj((B_ * SQ_) / 128, E_ / 128);       // (32, 8)
  gemm_nt_kernel<float, true><<<gproj, blk, 0, stream>>>(query, Wq, bq, (void*)Qh,
                                                         B_ * SQ_, E_, E_);
  gemm_nt_kernel<float, true><<<gproj, blk, 0, stream>>>(key_,  Wk, bk, (void*)Kh,
                                                         B_ * SK_, E_, E_);
  gemm_nt_kernel<float, true><<<gproj, blk, 0, stream>>>(value, Wv, bv, (void*)Vh,
                                                         B_ * SK_, E_, E_);

  dim3 gatt(SQ_ / 128, H_, B_);                 // (16, 16, 2)
  attn_scores_softmax_kernel<<<gatt, blk, 0, stream>>>(Qh, Kh, kpm, am, attnW);
  attn_pv_kernel<<<gatt, blk, 0, stream>>>(attnW, Vh, Ctx);

  gemm_nt_kernel<_Float16, false><<<gproj, blk, 0, stream>>>(Ctx, Wo, bo, d_out,
                                                             B_ * SQ_, E_, E_);
}
